// Block_22814866276680
// MI455X (gfx1250) — matrix operand order repaired and checked by reading, not compiled, not run on hardware
//
#include <hip/hip_runtime.h>
#include <math.h>
#include <stdint.h>

// ---------------------------------------------------------------------------
// Problem constants (from reference)
// ---------------------------------------------------------------------------
#define B_SZ   64
#define N_TOK  197
#define D_MOD  768
#define H_HD   12
#define DH     64
#define HID    3072
#define M_TOK  (B_SZ * N_TOK)          // 12608 = 788*16
#define NUM_REL 732
#define SPAD   224                     // padded token dim for P@V (7*32)

typedef __bf16 bf16_t;
typedef __attribute__((ext_vector_type(16))) __bf16 v16bf;
typedef __attribute__((ext_vector_type(8)))  float  v8f;

union Frag {
    uint4 u[2];
    v16bf v;
};

static __device__ inline v8f zero8() {
    v8f z = {0.f, 0.f, 0.f, 0.f, 0.f, 0.f, 0.f, 0.f};
    return z;
}

static __device__ inline v8f wmma_bf16(v16bf a, v16bf b, v8f c) {
    return __builtin_amdgcn_wmma_f32_16x16x32_bf16(false, a, false, b, (short)0, c,
                                                   false, false);
}

// A-matrix 16x32 bf16 fragment (ISA 7.12.2): lanes 0-15 hold M=lane,
// halves {K0..7, K16..23}; lanes 16-31 hold halves {K8..15, K24..31}.
static __device__ inline v16bf load_a_frag(const bf16_t* row, int laneHi) {
    Frag f;
    const bf16_t* p = row + laneHi * 8;
    f.u[0] = *(const uint4*)(p);
    f.u[1] = *(const uint4*)(p + 16);
    return f.v;
}

// B-matrix 32x16 bf16 fragment: lanes carry N; lanes 0-15 hold K0..7, K16..23,
// lanes 16-31 hold K8..15, K24..31 (same K order as the A fragment).
static __device__ inline v16bf load_b_frag(const bf16_t* row, int laneHi) {
    Frag f;
    const bf16_t* p = row + laneHi * 8;
    f.u[0] = *(const uint4*)(p);
    f.u[1] = *(const uint4*)(p + 16);
    return f.v;
}

// Async global -> LDS copy (CDNA5 Tensor/async data path, ASYNCcnt tracked).
// ldsAddr: LDS byte address (low 32 bits of generic shared pointer)
// voff:    per-lane byte offset added to the uniform 64-bit base (GVS mode)
static __device__ inline void async_copy_b128(unsigned ldsAddr, unsigned voff,
                                              const void* base, int ioff) {
    if (ioff == 16)
        asm volatile("global_load_async_to_lds_b128 %0, %1, %2 offset:16"
                     :
                     : "v"(ldsAddr), "v"(voff), "s"(base)
                     : "memory");
    else
        asm volatile("global_load_async_to_lds_b128 %0, %1, %2"
                     :
                     : "v"(ldsAddr), "v"(voff), "s"(base)
                     : "memory");
}

static __device__ inline void async_wait_all() {
    asm volatile("s_wait_asynccnt 0x0" ::: "memory");
}

// ---------------------------------------------------------------------------
// fp32 -> bf16 conversion (weights / activations)
// ---------------------------------------------------------------------------
__global__ void cvt_bf16_kernel(const float* __restrict__ in, bf16_t* __restrict__ out,
                                int n) {
    for (int i = blockIdx.x * blockDim.x + threadIdx.x; i < n;
         i += gridDim.x * blockDim.x)
        out[i] = (bf16_t)in[i];
}

// ---------------------------------------------------------------------------
// LayerNorm (row per block), fp32 in -> bf16 out
// ---------------------------------------------------------------------------
__global__ __launch_bounds__(256) void ln_kernel(const float* __restrict__ x,
                                                 const float* __restrict__ w,
                                                 const float* __restrict__ b,
                                                 bf16_t* __restrict__ out) {
    const int row = blockIdx.x;
    const float* xr = x + (size_t)row * D_MOD;
    float s = 0.f, s2 = 0.f;
    for (int i = threadIdx.x; i < D_MOD; i += 256) {
        float v = xr[i];
        s += v;
        s2 += v * v;
    }
    __shared__ float r1[256], r2[256];
    r1[threadIdx.x] = s;
    r2[threadIdx.x] = s2;
    __syncthreads();
    for (int st = 128; st > 0; st >>= 1) {
        if (threadIdx.x < st) {
            r1[threadIdx.x] += r1[threadIdx.x + st];
            r2[threadIdx.x] += r2[threadIdx.x + st];
        }
        __syncthreads();
    }
    const float mean = r1[0] * (1.f / D_MOD);
    const float var  = r2[0] * (1.f / D_MOD) - mean * mean;
    const float rs   = rsqrtf(var + 1e-5f);
    bf16_t* o = out + (size_t)row * D_MOD;
    for (int i = threadIdx.x; i < D_MOD; i += 256)
        o[i] = (bf16_t)((xr[i] - mean) * rs * w[i] + b[i]);
}

// ---------------------------------------------------------------------------
// WMMA GEMM: C[M, Nout] = A[M, K] @ W[Nout, K]^T   (bf16 in, fp32 acc)
// Block 256 thr = 8 waves (4 M x 2 N), block tile 64x128, wave tile 16x64.
// Tiles staged to LDS via async global->LDS (no VGPR round trip).
// MODE 0: QKV split epilogue   MODE 1: residual epilogue   MODE 2: GELU
// ---------------------------------------------------------------------------
template <int MODE>
__global__ __launch_bounds__(256) void gemm_bf16_kernel(
    const bf16_t* __restrict__ A, const bf16_t* __restrict__ Wm,
    int Nout, int K,
    const float* __restrict__ bias, const float* __restrict__ gamma,
    const float* __restrict__ resid, float* __restrict__ outF,
    bf16_t* __restrict__ outH, bf16_t* __restrict__ qkvO,
    bf16_t* __restrict__ vTO) {
    __shared__ bf16_t sA[64 * 32];      // 4 KB
    __shared__ bf16_t sW[128 * 32];     // 8 KB

    const int tid = threadIdx.x;
    const int lane = tid & 31, wave = tid >> 5;
    const int laneN = lane & 15, laneHi = lane >> 4;
    const int waveM = wave >> 1, waveN = wave & 1;
    const int mBlock = blockIdx.x * 64;
    const int nBlock = blockIdx.y * 128;

    v8f acc[4];
#pragma unroll
    for (int t = 0; t < 4; ++t) acc[t] = zero8();

    const int rowA = tid >> 2;          // 0..63
    const int chkA = (tid & 3) * 8;     // halves 0,8,16,24
    const int rowW = tid >> 1;          // 0..127
    const int chkW = (tid & 1) * 16;    // halves 0,16
    const int ktMax = K >> 5;

    const unsigned ldsA = (unsigned)(uintptr_t)(&sA[rowA * 32 + chkA]);
    const unsigned ldsW = (unsigned)(uintptr_t)(&sW[rowW * 32 + chkW]);

    for (int kt = 0; kt < ktMax; ++kt) {
        const int k0 = kt << 5;
        const unsigned vofA =
            (unsigned)(((size_t)(mBlock + rowA) * K + k0 + chkA) * sizeof(bf16_t));
        const unsigned vofW =
            (unsigned)(((size_t)(nBlock + rowW) * K + k0 + chkW) * sizeof(bf16_t));
        async_copy_b128(ldsA, vofA, A, 0);
        async_copy_b128(ldsW, vofW, Wm, 0);
        async_copy_b128(ldsW, vofW, Wm, 16);
        if (kt + 1 < ktMax) {
            __builtin_prefetch(&A[(size_t)(mBlock + rowA) * K + k0 + 32 + chkA], 0, 1);
            __builtin_prefetch(&Wm[(size_t)(nBlock + rowW) * K + k0 + 32 + chkW], 0, 1);
        }
        async_wait_all();
        __syncthreads();
        v16bf a = load_a_frag(&sA[(waveM * 16 + laneN) * 32], laneHi);
#pragma unroll
        for (int t = 0; t < 4; ++t) {
            v16bf bt = load_b_frag(&sW[(waveN * 64 + t * 16 + laneN) * 32], laneHi);
            acc[t] = wmma_bf16(a, bt, acc[t]);
        }
        __syncthreads();
    }

#pragma unroll
    for (int t = 0; t < 4; ++t) {
#pragma unroll
        for (int r = 0; r < 8; ++r) {
            const int m = mBlock + waveM * 16 + r + laneHi * 8;
            const int c = nBlock + waveN * 64 + t * 16 + laneN;
            const float val = acc[t][r];
            if (MODE == 0) {
                // qkv split: bias = q_bias, gamma = v_bias
                const int which = c / D_MOD;
                const int d     = c - which * D_MOD;
                const int head  = d >> 6;
                const int dh    = d & 63;
                const int bidx  = m / N_TOK;
                const int n     = m - bidx * N_TOK;
                float bb = (which == 0) ? bias[d] : ((which == 2) ? gamma[d] : 0.f);
                float vv = val + bb;
                if (which == 0) vv *= 0.125f;  // DH^-0.5
                qkvO[(size_t)((which * B_SZ + bidx) * H_HD + head) * (N_TOK * DH) +
                     n * DH + dh] = (bf16_t)vv;
                if (which == 2)
                    vTO[(size_t)((bidx * H_HD + head) * DH + dh) * SPAD + n] =
                        (bf16_t)vv;
            } else if (MODE == 1) {
                // out = resid + gamma * (acc + bias)
                const size_t idx = (size_t)m * Nout + c;
                outF[idx] = resid[idx] + gamma[c] * (val + bias[c]);
            } else {
                // exact GELU
                float u = val + bias[c];
                outH[(size_t)m * Nout + c] =
                    (bf16_t)(0.5f * u * (1.f + erff(u * 0.70710678118654752f)));
            }
        }
    }
}

// ---------------------------------------------------------------------------
// Attention: per (b,h) block, 4 waves, each wave owns 16-row query tiles.
// S = Q@K^T (+rpb, mask) -> per-wave LDS staging -> softmax -> O = P@V.
// ---------------------------------------------------------------------------
__global__ __launch_bounds__(128) void attn_kernel(
    const bf16_t* __restrict__ qkv, const bf16_t* __restrict__ vT,
    const float* __restrict__ rpb, const int* __restrict__ relidx,
    bf16_t* __restrict__ ctx) {
    __shared__ float stag[4 * 16 * SPAD];   // 57344 B
    __shared__ float rowinv[4 * 16];

    const int bh = blockIdx.x;
    const int b = bh / H_HD, h = bh % H_HD;
    const int lane = threadIdx.x & 31, wave = threadIdx.x >> 5;
    const int laneN = lane & 15, laneHi = lane >> 4;

    const bf16_t* qb = qkv + (size_t)(b * H_HD + h) * (N_TOK * DH);
    const bf16_t* kb = qkv + (size_t)((B_SZ + b) * H_HD + h) * (N_TOK * DH);
    const bf16_t* vt = vT + (size_t)(b * H_HD + h) * DH * SPAD;
    float* st = stag + wave * 16 * SPAD;

    for (int mt = wave; mt < 13; mt += 4) {
        const int m0 = mt * 16;
        int rq = m0 + laneN;
        if (rq > N_TOK - 1) rq = N_TOK - 1;  // clamped rows masked later
        Frag qa0, qa1;
        {
            const bf16_t* p = qb + rq * DH + laneHi * 8;
            qa0.u[0] = *(const uint4*)(p);
            qa0.u[1] = *(const uint4*)(p + 16);
            qa1.u[0] = *(const uint4*)(p + 32);
            qa1.u[1] = *(const uint4*)(p + 48);
        }
        // ---- S = Q @ K^T, +rpb (branchless clamped gather), mask, stage ----
        for (int jt = 0; jt < 13; ++jt) {
            int nk = jt * 16 + laneN;
            if (nk > N_TOK - 1) nk = N_TOK - 1;
            Frag kb0, kb1;
            {
                const bf16_t* p = kb + nk * DH + laneHi * 8;
                kb0.u[0] = *(const uint4*)(p);
                kb0.u[1] = *(const uint4*)(p + 16);
                kb1.u[0] = *(const uint4*)(p + 32);
                kb1.u[1] = *(const uint4*)(p + 48);
            }
            v8f s = zero8();
            s = wmma_bf16(qa0.v, kb0.v, s);
            s = wmma_bf16(qa1.v, kb1.v, s);
            const int n  = jt * 16 + laneN;
            const int nc = (n < N_TOK) ? n : (N_TOK - 1);
#pragma unroll
            for (int r = 0; r < 8; ++r) {
                const int mrow = r + laneHi * 8;
                const int m  = m0 + mrow;
                const int mc = (m < N_TOK) ? m : (N_TOK - 1);
                const int idx = relidx[mc * N_TOK + nc];      // always in-bounds
                const float biased = s[r] + rpb[idx * H_HD + h];
                const float val = (m < N_TOK && n < N_TOK) ? biased : -3.0e38f;
                st[mrow * SPAD + jt * 16 + laneN] = val;
            }
        }
        // fill padded cols 208..223
        for (int i = lane; i < 256; i += 32)
            st[(i >> 4) * SPAD + 208 + (i & 15)] = -3.0e38f;

        // ---- softmax over 224 cols (lane halves split columns, float4) ----
        {
            float4* rp4 = (float4*)(st + laneN * SPAD + laneHi * 112);
            float mx = -3.4e38f;
            for (int c = 0; c < 28; ++c) {
                float4 v = rp4[c];
                mx = fmaxf(mx, fmaxf(fmaxf(v.x, v.y), fmaxf(v.z, v.w)));
            }
            mx = fmaxf(mx, __shfl_xor(mx, 16));
            const bool ok = mx > -1e30f;  // all-masked (padded) rows -> P = 0
            float sum = 0.f;
            for (int c = 0; c < 28; ++c) {
                float4 v = rp4[c];
                float4 e;
                e.x = ok ? __expf(v.x - mx) : 0.f;
                e.y = ok ? __expf(v.y - mx) : 0.f;
                e.z = ok ? __expf(v.z - mx) : 0.f;
                e.w = ok ? __expf(v.w - mx) : 0.f;
                rp4[c] = e;
                sum += e.x + e.y + e.z + e.w;
            }
            sum += __shfl_xor(sum, 16);
            if (lane < 16) rowinv[wave * 16 + laneN] = 1.f / fmaxf(sum, 1e-20f);
        }

        // ---- O = P @ V (vT rows are K-contiguous, padded cols are zero) ----
        v8f o[4];
#pragma unroll
        for (int nt = 0; nt < 4; ++nt) o[nt] = zero8();
        for (int kt = 0; kt < 7; ++kt) {
            const int k0 = kt * 32;
            const float4* p4 = (const float4*)(st + laneN * SPAD + k0 + laneHi * 8);
            float4 a0 = p4[0], a1 = p4[1];       // halves 0..7
            float4 c0 = p4[4], c1 = p4[5];       // halves +16 .. +23
            float vals[16] = {a0.x, a0.y, a0.z, a0.w, a1.x, a1.y, a1.z, a1.w,
                              c0.x, c0.y, c0.z, c0.w, c1.x, c1.y, c1.z, c1.w};
            Frag pf;
#pragma unroll
            for (int i = 0; i < 16; ++i) pf.v[i] = (bf16_t)vals[i];
#pragma unroll
            for (int nt = 0; nt < 4; ++nt) {
                const bf16_t* vr = vt + (nt * 16 + laneN) * SPAD + k0 + laneHi * 8;
                Frag vf;
                vf.u[0] = *(const uint4*)(vr);
                vf.u[1] = *(const uint4*)(vr + 16);
                o[nt] = wmma_bf16(pf.v, vf.v, o[nt]);
            }
        }
#pragma unroll
        for (int nt = 0; nt < 4; ++nt) {
#pragma unroll
            for (int r = 0; r < 8; ++r) {
                const int mrow = r + laneHi * 8;
                const int m = m0 + mrow;
                if (m < N_TOK) {
                    const float rv = rowinv[wave * 16 + mrow];
                    ctx[(size_t)(b * N_TOK + m) * D_MOD + h * DH + nt * 16 + laneN] =
                        (bf16_t)(o[nt][r] * rv);
                }
            }
        }
    }
}

// ---------------------------------------------------------------------------
// Workspace layout (bytes, all offsets 256-aligned)
// ---------------------------------------------------------------------------
#define OFF_H1   ((size_t)0)                              // 19,365,888 (bf16 h / h2)
#define OFF_QKV  ((size_t)19365888)                       // 58,097,664 (bf16 qkv / mid alias)
#define OFF_VT   (OFF_QKV + (size_t)58097664)             // 22,020,096 (bf16 vT)
#define OFF_CTX  (OFF_VT + (size_t)22020096)              // 19,365,888 (bf16 ctx)
#define OFF_X1   (OFF_CTX + (size_t)19365888)             // 38,731,776 (f32 x1)
#define OFF_WQKV (OFF_X1 + (size_t)38731776)              //  3,538,944
#define OFF_WPRJ (OFF_WQKV + (size_t)3538944)             //  1,179,648
#define OFF_WFC1 (OFF_WPRJ + (size_t)1179648)             //  4,718,592
#define OFF_WFC2 (OFF_WFC1 + (size_t)4718592)             //  4,718,592
#define VT_BYTES ((size_t)22020096)

extern "C" void kernel_launch(void* const* d_in, const int* in_sizes, int n_in,
                              void* d_out, int out_size, void* d_ws, size_t ws_size,
                              hipStream_t stream) {
    const float* x       = (const float*)d_in[0];
    const float* norm1_w = (const float*)d_in[1];
    const float* norm1_b = (const float*)d_in[2];
    const float* qkv_w   = (const float*)d_in[3];
    const float* q_bias  = (const float*)d_in[4];
    const float* v_bias  = (const float*)d_in[5];
    const float* rpb     = (const float*)d_in[6];
    const float* proj_w  = (const float*)d_in[7];
    const float* proj_b  = (const float*)d_in[8];
    const float* gamma1  = (const float*)d_in[9];
    const float* norm2_w = (const float*)d_in[10];
    const float* norm2_b = (const float*)d_in[11];
    const float* fc1_w   = (const float*)d_in[12];
    const float* fc1_b   = (const float*)d_in[13];
    const float* fc2_w   = (const float*)d_in[14];
    const float* fc2_b   = (const float*)d_in[15];
    const float* gamma2  = (const float*)d_in[16];
    const int*   relidx  = (const int*)d_in[17];
    float* out = (float*)d_out;

    char* ws = (char*)d_ws;
    bf16_t* h1   = (bf16_t*)(ws + OFF_H1);
    bf16_t* qkvB = (bf16_t*)(ws + OFF_QKV);
    bf16_t* vTB  = (bf16_t*)(ws + OFF_VT);
    bf16_t* ctxB = (bf16_t*)(ws + OFF_CTX);
    float*  x1   = (float*)(ws + OFF_X1);
    bf16_t* wqkv = (bf16_t*)(ws + OFF_WQKV);
    bf16_t* wprj = (bf16_t*)(ws + OFF_WPRJ);
    bf16_t* wfc1 = (bf16_t*)(ws + OFF_WFC1);
    bf16_t* wfc2 = (bf16_t*)(ws + OFF_WFC2);
    bf16_t* h2   = h1;                       // alias: h1 dead after QKV GEMM
    bf16_t* midB = qkvB;                     // alias: qkv+vT dead after attention

    // zero vT so padded token columns (197..223) feed zeros into WMMA
    hipMemsetAsync(vTB, 0, VT_BYTES, stream);

    // weight conversions fp32 -> bf16
    cvt_bf16_kernel<<<2048, 256, 0, stream>>>(qkv_w, wqkv, 3 * D_MOD * D_MOD);
    cvt_bf16_kernel<<<2048, 256, 0, stream>>>(proj_w, wprj, D_MOD * D_MOD);
    cvt_bf16_kernel<<<2048, 256, 0, stream>>>(fc1_w, wfc1, HID * D_MOD);
    cvt_bf16_kernel<<<2048, 256, 0, stream>>>(fc2_w, wfc2, D_MOD * HID);

    // LN1
    ln_kernel<<<M_TOK, 256, 0, stream>>>(x, norm1_w, norm1_b, h1);

    // QKV GEMM: [12608,768] x [2304,768]^T
    gemm_bf16_kernel<0><<<dim3(M_TOK / 64, (3 * D_MOD) / 128), 256, 0, stream>>>(
        h1, wqkv, 3 * D_MOD, D_MOD, q_bias, v_bias, nullptr, nullptr, nullptr,
        qkvB, vTB);

    // attention
    attn_kernel<<<B_SZ * H_HD, 128, 0, stream>>>(qkvB, vTB, rpb, relidx, ctxB);

    // proj + residual: x1 = x + gamma1 * (ctx @ proj_w^T + proj_b)
    gemm_bf16_kernel<1><<<dim3(M_TOK / 64, D_MOD / 128), 256, 0, stream>>>(
        ctxB, wprj, D_MOD, D_MOD, proj_b, gamma1, x, x1, nullptr, nullptr, nullptr);

    // LN2
    ln_kernel<<<M_TOK, 256, 0, stream>>>(x1, norm2_w, norm2_b, h2);

    // FC1 + GELU
    gemm_bf16_kernel<2><<<dim3(M_TOK / 64, HID / 128), 256, 0, stream>>>(
        h2, wfc1, HID, D_MOD, fc1_b, nullptr, nullptr, nullptr, midB, nullptr,
        nullptr);

    // FC2 + residual -> out
    gemm_bf16_kernel<1><<<dim3(M_TOK / 64, D_MOD / 128), 256, 0, stream>>>(
        midB, wfc2, D_MOD, HID, fc2_b, gamma2, x1, out, nullptr, nullptr, nullptr);
}
